// SelfAttention_43937515438128
// MI455X (gfx1250) — matrix operand order repaired and checked by reading, not compiled, not run on hardware
//
#include <hip/hip_runtime.h>

typedef __attribute__((ext_vector_type(16))) __bf16 v16bf;
typedef __attribute__((ext_vector_type(8)))  float  v8f;
typedef unsigned int u32x4 __attribute__((ext_vector_type(4)));
typedef int          i32x8 __attribute__((ext_vector_type(8)));
typedef int          i32x4 __attribute__((ext_vector_type(4)));

#define B_  8
#define C_  128
#define N_  4096
#define D_  16

#if __has_builtin(__builtin_amdgcn_tensor_load_to_lds) && \
    __has_builtin(__builtin_amdgcn_s_wait_tensorcnt)
#define HAVE_TDM 1
#else
#define HAVE_TDM 0
#endif

__device__ __forceinline__ unsigned short f2bf(float f) {
  unsigned u = __builtin_bit_cast(unsigned, f);
  return (unsigned short)((u + 0x7fffu + ((u >> 16) & 1u)) >> 16);
}
__device__ __forceinline__ unsigned pack2(float a, float b) {
  return (unsigned)f2bf(a) | ((unsigned)f2bf(b) << 16);
}

union FragU { v16bf v; unsigned u[8]; };

// A-matrix fragment (16x32 bf16) from row-major [row][K] LDS tile.
__device__ __forceinline__ v16bf load_frag_a(const unsigned short* base, int stride,
                                             int row0, int k0) {
  int lane = threadIdx.x & 31;
  const unsigned short* p = base + (row0 + (lane & 15)) * stride + k0 + ((lane >> 4) << 3);
  FragU f;
#pragma unroll
  for (int i = 0; i < 8; ++i) {
    int k = ((i >> 2) << 4) + ((i & 3) << 1);
    f.u[i] = *(const unsigned*)(p + k);
  }
  return f.v;
}
// B-matrix fragment (32x16 bf16) from row-major [col][K] LDS tile.
__device__ __forceinline__ v16bf load_frag_b(const unsigned short* base, int stride,
                                             int col0, int k0) {
  int lane = threadIdx.x & 31;
  const unsigned short* p = base + (col0 + (lane & 15)) * stride + k0 + ((lane >> 4) << 3);
  FragU f;
#pragma unroll
  for (int i = 0; i < 8; ++i) f.u[i] = *(const unsigned*)(p + ((i >> 2) << 4) + ((i & 3) << 1));
  return f.v;
}

__device__ __forceinline__ v8f wmma_bf16(v16bf a, v16bf b, v8f c) {
  return __builtin_amdgcn_wmma_f32_16x16x32_bf16(false, a, false, b, (short)0, c,
                                                 false, false);
}

#if HAVE_TDM
// TDM load of a 2D tile (bf16 elements) into LDS, with per-row LDS padding.
// dim0/stride0/tile0 etc. in elements; pad codes per D# spec.
__device__ __forceinline__ void tdm_load_2d(unsigned lds_off, const void* gptr,
                                            int dim0, long long stride0, int dim1,
                                            int tile0, int tile1, int ivl_code,
                                            int amt_code) {
  unsigned long long ga = (unsigned long long)(size_t)gptr;
  u32x4 g0;
  g0[0] = 1u;  // count=1, user descriptor
  g0[1] = lds_off;
  g0[2] = (unsigned)ga;
  g0[3] = ((unsigned)(ga >> 32) & 0x01FFFFFFu) | 0x80000000u;  // addr hi | type=2
  i32x8 g1;
  g1[0] = (1 << 16) | (1 << 20) | (ivl_code << 22) | (amt_code << 25);  // 2B, pad_en
  g1[1] = (int)((dim0 & 0xFFFF) << 16);                     // tensor_dim0 lo16
  g1[2] = (int)(((unsigned)dim0 >> 16) | ((dim1 & 0xFFFF) << 16));
  g1[3] = (int)(((unsigned)dim1 >> 16) | ((unsigned)tile0 << 16));
  g1[4] = tile1 & 0xFFFF;                                   // tile_dim1, tile_dim2=0
  g1[5] = (int)(stride0 & 0xFFFFFFFFll);                    // dim0_stride lo32
  g1[6] = (int)((stride0 >> 32) & 0xFFFFll);                // dim0_stride hi16
  g1[7] = 0;
  i32x4 z4 = {0, 0, 0, 0};
  i32x8 z8 = {0, 0, 0, 0, 0, 0, 0, 0};
  __builtin_amdgcn_tensor_load_to_lds(g0, g1, z4, z4, z8, 0);
}
#endif

// ---------------- K1: GroupNorm statistics (mean, rsqrt(var+eps)) ----------------
__global__ __launch_bounds__(256) void gn_stats_kernel(const float* __restrict__ x,
                                                       float* __restrict__ stats) {
  int bg = blockIdx.x;                         // b*32 + g
  const float* base = x + (size_t)bg * 16384;  // 4 channels * 4096, contiguous
  int t = threadIdx.x;
  float s1 = 0.f, s2 = 0.f;
#pragma unroll
  for (int s = 0; s < 16; ++s) {
    float4 v = *(const float4*)(base + (size_t)(t + s * 256) * 4);
    s1 += v.x + v.y + v.z + v.w;
    s2 += v.x * v.x + v.y * v.y + v.z * v.z + v.w * v.w;
  }
#pragma unroll
  for (int m = 1; m < 32; m <<= 1) {
    s1 += __shfl_xor(s1, m, 32);
    s2 += __shfl_xor(s2, m, 32);
  }
  __shared__ float r1[8], r2[8];
  int wid = t >> 5, lane = t & 31;
  if (lane == 0) { r1[wid] = s1; r2[wid] = s2; }
  __syncthreads();
  if (t == 0) {
    float a = 0.f, b = 0.f;
#pragma unroll
    for (int i = 0; i < 8; ++i) { a += r1[i]; b += r2[i]; }
    float mu = a * (1.f / 16384.f);
    float var = b * (1.f / 16384.f) - mu * mu;
    stats[bg * 2 + 0] = mu;
    stats[bg * 2 + 1] = rsqrtf(var + 1e-5f);
  }
}

// ---------------- K2: fused GroupNorm-apply + QKV projection GEMM ----------------
// q/k: D[pos][feat] (A = act tile, B = weights).  v: D[c_out][pos] (A = v_w, B = act
// tile) so v lands in [B][C][N] layout with coalesced stores and K3 needs no transpose.
__global__ __launch_bounds__(256) void qkv_kernel(
    const float* __restrict__ x, const float* __restrict__ stats,
    const float* __restrict__ gn_w, const float* __restrict__ gn_b,
    const float* __restrict__ q_w, const float* __restrict__ q_b,
    const float* __restrict__ k_w, const float* __restrict__ k_b,
    const float* __restrict__ v_w, const float* __restrict__ v_b,
    unsigned short* __restrict__ qb, unsigned short* __restrict__ kb,
    unsigned short* __restrict__ vb) {
  const int KS = 136;
  __shared__ unsigned short at[64 * 136];   // normalized activations [pos][ch]
  __shared__ unsigned short wt[160 * 136];  // q_w|k_w|v_w stacked   [out][ch]
  int b = blockIdx.x >> 6;
  int p0 = (blockIdx.x & 63) * 64;
  int t = threadIdx.x;

#pragma unroll
  for (int s = 0; s < 8; ++s) {
    int flat = (t + s * 256) * 4;
    int c = flat >> 6;
    int m0 = flat & 63;
    float4 xv = *(const float4*)(x + ((size_t)(b * C_ + c)) * N_ + p0 + m0);
    float mu = stats[((b << 5) + (c >> 2)) * 2 + 0];
    float rs = stats[((b << 5) + (c >> 2)) * 2 + 1];
    float w = gn_w[c], bi = gn_b[c];
    at[(m0 + 0) * KS + c] = f2bf((xv.x - mu) * rs * w + bi);
    at[(m0 + 1) * KS + c] = f2bf((xv.y - mu) * rs * w + bi);
    at[(m0 + 2) * KS + c] = f2bf((xv.z - mu) * rs * w + bi);
    at[(m0 + 3) * KS + c] = f2bf((xv.w - mu) * rs * w + bi);
  }
#pragma unroll
  for (int s = 0; s < 20; ++s) {
    int flat = (t + s * 256) * 4;
    int row = flat >> 7;
    int c0 = flat & 127;
    const float* src = (row < 16)   ? q_w + row * C_ + c0
                       : (row < 32) ? k_w + (row - 16) * C_ + c0
                                    : v_w + (row - 32) * C_ + c0;
    float4 wv = *(const float4*)src;
    *(uint2*)(&wt[row * KS + c0]) = make_uint2(pack2(wv.x, wv.y), pack2(wv.z, wv.w));
  }
  __syncthreads();

  int wid = t >> 5, lane = t & 31;
  v8f zf = {0.f, 0.f, 0.f, 0.f, 0.f, 0.f, 0.f, 0.f};
  v8f acc[5];
#pragma unroll
  for (int i = 0; i < 5; ++i) acc[i] = zf;
  // 40 tile-jobs: 0..31 = v (8 row-tiles x 4 pos-tiles); 32..39 = q/k (4 x 2).
  // Branch-free descriptors (wave-uniform) so fragment loads stay unconditional.
  const unsigned short* abase[5];
  const unsigned short* bbase[5];
  int arow[5], bcol[5];
#pragma unroll
  for (int s = 0; s < 5; ++s) {
    int j = wid + 8 * s;
    bool isv = (j < 32);
    int jj = j - 32;
    abase[s] = isv ? wt : at;
    arow[s]  = isv ? (32 + (j >> 2) * 16) : ((jj >> 1) * 16);
    bbase[s] = isv ? at : wt;
    bcol[s]  = isv ? ((j & 3) * 16) : ((jj & 1) * 16);
  }
#pragma unroll
  for (int kk = 0; kk < 4; ++kk) {
#pragma unroll
    for (int s = 0; s < 5; ++s) {
      v16bf a  = load_frag_a(abase[s], KS, arow[s], kk * 32);
      v16bf bf = load_frag_b(bbase[s], KS, bcol[s], kk * 32);
      acc[s] = wmma_bf16(a, bf, acc[s]);
    }
  }
#pragma unroll
  for (int s = 0; s < 5; ++s) {
    int j = wid + 8 * s;
    if (j < 32) {
      int pos = p0 + (j & 3) * 16 + (lane & 15);
#pragma unroll
      for (int r = 0; r < 8; ++r) {
        int c = (j >> 2) * 16 + ((lane >> 4) << 3) + r;
        vb[((size_t)b * C_ + c) * N_ + pos] = f2bf(acc[s][r] + v_b[c]);
      }
    } else {
      int jj = j - 32;
      int col = (jj & 1) * 16 + (lane & 15);
#pragma unroll
      for (int r = 0; r < 8; ++r) {
        int m = (jj >> 1) * 16 + ((lane >> 4) << 3) + r;
        size_t pos = (size_t)b * N_ + p0 + m;
        if (col < 16) qb[pos * D_ + col]      = f2bf(acc[s][r] + q_b[col]);
        else          kb[pos * D_ + col - 16] = f2bf(acc[s][r] + k_b[col - 16]);
      }
    }
  }
}

// ---------------- K3: flash attention, TDM double-buffered V tiles ----------------
__global__ __launch_bounds__(256) void attn_kernel(const unsigned short* __restrict__ qb,
                                                   const unsigned short* __restrict__ kb,
                                                   const unsigned short* __restrict__ vb,
                                                   unsigned short* __restrict__ ab) {
  // pool: kt[64][32] | vt[2][128][72] | (qs[128][32] aliased with ps[8][16][72])
  __shared__ __align__(16) unsigned short pool[2048 + 2 * 9216 + 9216];
  unsigned short* kt  = pool;
  unsigned short* vt0 = pool + 2048;
  unsigned short* vt1 = pool + 2048 + 9216;
  unsigned short* qs  = pool + 2048 + 2 * 9216;  // dead after prologue
  unsigned short* ps  = pool + 2048 + 2 * 9216;  // reuses qs space

  int b = blockIdx.x >> 5;
  int q0 = (blockIdx.x & 31) * 128;
  int t = threadIdx.x, wid = t >> 5, lane = t & 31;

  // keep LLVM from proving the TDM-written arrays store-free
  if ((int)blockIdx.x < 0) { vt0[t] = 0; vt1[t] = 0; }

  {  // zero K-dim pad columns (kt pads survive all iterations; TDM skips them)
    int flat = t * 8;
    *(uint4*)(&qs[(flat >> 4) * 32 + 16 + (flat & 15)]) = make_uint4(0, 0, 0, 0);
    if (t < 128) {
      int f2 = t * 8;
      *(uint4*)(&kt[(f2 >> 4) * 32 + 16 + (f2 & 15)]) = make_uint4(0, 0, 0, 0);
    }
  }
  {  // stage Q once
    int flat = t * 8;
    int m = flat >> 4, c0 = flat & 15;
    *(uint4*)(&qs[m * 32 + c0]) =
        *(const uint4*)(qb + ((size_t)b * N_ + q0 + m) * D_ + c0);
  }
  uint2 kreg;  // software-pipelined K tile (tile 0)
  {
    int flat = t * 4;
    int j = flat >> 4, c0 = flat & 15;
    kreg = *(const uint2*)(kb + ((size_t)b * N_ + j) * D_ + c0);
  }
#if HAVE_TDM
  if (wid == 0)  // V tile 0 -> vt0
    tdm_load_2d((unsigned)(size_t)vt0, vb + (size_t)b * C_ * N_, N_, N_, C_, 64, C_,
                /*ivl=32dw*/ 4, /*amt=4dw*/ 3);
#endif
  __syncthreads();

  v16bf qfrag = load_frag_a(qs, 32, wid * 16, 0);  // qs is dead after this

  v8f zf = {0.f, 0.f, 0.f, 0.f, 0.f, 0.f, 0.f, 0.f};
  v8f acc[8];
#pragma unroll
  for (int i = 0; i < 8; ++i) acc[i] = zf;
  float rmax[8], rsum[8];
#pragma unroll
  for (int r = 0; r < 8; ++r) { rmax[r] = -1e30f; rsum[r] = 0.f; }

  unsigned short* pwv = ps + wid * 16 * 72;
  const int nt = N_ / 64;

  for (int it = 0; it < nt; ++it) {
    int j0 = it * 64;
    {  // commit pipelined K tile, prefetch next into registers
      int flat = t * 4;
      int j = flat >> 4, c0 = flat & 15;
      *(uint2*)(&kt[j * 32 + c0]) = kreg;
      if (it + 1 < nt)
        kreg = *(const uint2*)(kb + ((size_t)b * N_ + j0 + 64 + j) * D_ + c0);
    }
#if HAVE_TDM
    if (wid == 0) {
      if (it + 1 < nt) {  // async-load next V tile into the other buffer
        tdm_load_2d((unsigned)(size_t)((it & 1) ? vt0 : vt1),
                    vb + (size_t)b * C_ * N_ + j0 + 64, N_, N_, C_, 64, C_, 4, 3);
        __builtin_amdgcn_s_wait_tensorcnt(1);  // newest may stay in flight
      } else {
        __builtin_amdgcn_s_wait_tensorcnt(0);  // drain for the last tile
      }
      __asm__ volatile("" ::: "memory");
    }
    const unsigned short* vcur = (it & 1) ? vt1 : vt0;
#else
#pragma unroll
    for (int s = 0; s < 4; ++s) {  // cooperative copy fallback
      int flat = (t + s * 256) * 8;
      int c = flat >> 6, jj = flat & 63;
      *(uint4*)(&vt0[c * 72 + jj]) =
          *(const uint4*)(vb + ((size_t)b * C_ + c) * N_ + j0 + jj);
    }
    const unsigned short* vcur = vt0;
#endif
    __syncthreads();

    // S = Q * K^T
    float sv[4][8];
#pragma unroll
    for (int tt = 0; tt < 4; ++tt) {
      v16bf bf = load_frag_b(kt, 32, tt * 16, 0);
      v8f s = wmma_bf16(qfrag, bf, zf);
#pragma unroll
      for (int r = 0; r < 8; ++r) sv[tt][r] = s[r] * 0.25f;  // d^-1/2
    }
    // online softmax
    float tmax[8], tsum[8], corr[8];
#pragma unroll
    for (int r = 0; r < 8; ++r) {
      float mx = sv[0][r];
#pragma unroll
      for (int tt = 1; tt < 4; ++tt) mx = fmaxf(mx, sv[tt][r]);
      tmax[r] = mx;
    }
#pragma unroll
    for (int m = 1; m < 16; m <<= 1)
#pragma unroll
      for (int r = 0; r < 8; ++r) tmax[r] = fmaxf(tmax[r], __shfl_xor(tmax[r], m, 32));
#pragma unroll
    for (int r = 0; r < 8; ++r) {
      float nm = fmaxf(rmax[r], tmax[r]);
      corr[r] = __expf(rmax[r] - nm);
      rmax[r] = nm;
      tsum[r] = 0.f;
    }
#pragma unroll
    for (int tt = 0; tt < 4; ++tt)
#pragma unroll
      for (int r = 0; r < 8; ++r) {
        float p = __expf(sv[tt][r] - rmax[r]);
        sv[tt][r] = p;
        tsum[r] += p;
      }
#pragma unroll
    for (int m = 1; m < 16; m <<= 1)
#pragma unroll
      for (int r = 0; r < 8; ++r) tsum[r] += __shfl_xor(tsum[r], m, 32);
#pragma unroll
    for (int r = 0; r < 8; ++r) rsum[r] = rsum[r] * corr[r] + tsum[r];
#pragma unroll
    for (int ct = 0; ct < 8; ++ct)
#pragma unroll
      for (int r = 0; r < 8; ++r) acc[ct][r] *= corr[r];
    // P: C-layout -> A-layout via per-wave LDS patch
#pragma unroll
    for (int tt = 0; tt < 4; ++tt)
#pragma unroll
      for (int r = 0; r < 8; ++r)
        pwv[(((lane >> 4) << 3) + r) * 72 + tt * 16 + (lane & 15)] = f2bf(sv[tt][r]);
    // acc += P * V
#pragma unroll
    for (int kk = 0; kk < 2; ++kk) {
      v16bf af = load_frag_a(pwv, 72, 0, kk * 32);
#pragma unroll
      for (int ct = 0; ct < 8; ++ct) {
        v16bf bf = load_frag_b(vcur, 72, ct * 16, kk * 32);
        acc[ct] = wmma_bf16(af, bf, acc[ct]);
      }
    }
    __syncthreads();
  }
#pragma unroll
  for (int ct = 0; ct < 8; ++ct)
#pragma unroll
    for (int r = 0; r < 8; ++r) {
      int m = ((lane >> 4) << 3) + r;
      size_t pos = (size_t)b * N_ + q0 + wid * 16 + m;
      ab[pos * C_ + ct * 16 + (lane & 15)] = f2bf(acc[ct][r] / rsum[r]);
    }
}

// ---------------- K4: output projection + bias + residual ----------------
__global__ __launch_bounds__(256) void proj_kernel(const float* __restrict__ x,
                                                   const float* __restrict__ p_w,
                                                   const float* __restrict__ p_b,
                                                   const unsigned short* __restrict__ ab,
                                                   float* __restrict__ out) {
  const int KS = 136;
  __shared__ unsigned short pw[128 * 136];  // p_w [c_out][c_in] bf16
  __shared__ unsigned short bt[64 * 136];   // a^T  [pos][c_in]  bf16
  int b = blockIdx.x >> 6;
  int p0 = (blockIdx.x & 63) * 64;
  int t = threadIdx.x;
  if ((int)blockIdx.x < 0) bt[t] = 0;  // see K3 note
#if HAVE_TDM
  if (t < 32)  // TDM the a^T tile while all threads convert p_w below
    tdm_load_2d((unsigned)(size_t)bt, ab + ((size_t)b * N_ + p0) * C_, C_, C_, 64, C_,
                64, /*ivl=64dw*/ 5, /*amt=4dw*/ 3);
#endif
#pragma unroll
  for (int s = 0; s < 16; ++s) {
    int flat = (t + s * 256) * 4;
    int row = flat >> 7, c0 = flat & 127;
    float4 wv = *(const float4*)(p_w + row * C_ + c0);
    *(uint2*)(&pw[row * KS + c0]) = make_uint2(pack2(wv.x, wv.y), pack2(wv.z, wv.w));
  }
#if HAVE_TDM
  if (t < 32) {
    __builtin_amdgcn_s_wait_tensorcnt(0);
    __asm__ volatile("" ::: "memory");
  }
#else
#pragma unroll
  for (int s = 0; s < 4; ++s) {
    int flat = (t + s * 256) * 8;
    int m = flat >> 7, c0 = flat & 127;
    *(uint4*)(&bt[m * KS + c0]) =
        *(const uint4*)(ab + ((size_t)b * N_ + p0 + m) * C_ + c0);
  }
#endif
  __syncthreads();
  int wid = t >> 5, lane = t & 31;
  int m0 = wid * 16;
  v8f zf = {0.f, 0.f, 0.f, 0.f, 0.f, 0.f, 0.f, 0.f};
  v8f acc[4];
#pragma unroll
  for (int i = 0; i < 4; ++i) acc[i] = zf;
#pragma unroll
  for (int kk = 0; kk < 4; ++kk) {
    v16bf a = load_frag_a(pw, KS, m0, kk * 32);
#pragma unroll
    for (int ct = 0; ct < 4; ++ct) {
      v16bf bf = load_frag_b(bt, KS, ct * 16, kk * 32);
      acc[ct] = wmma_bf16(a, bf, acc[ct]);
    }
  }
#pragma unroll
  for (int ct = 0; ct < 4; ++ct)
#pragma unroll
    for (int r = 0; r < 8; ++r) {
      int c = m0 + ((lane >> 4) << 3) + r;
      int pos = p0 + ct * 16 + (lane & 15);
      size_t idx = ((size_t)(b * C_ + c)) * N_ + pos;
      out[idx] = x[idx] + acc[ct][r] + p_b[c];
    }
}

extern "C" void kernel_launch(void* const* d_in, const int* in_sizes, int n_in,
                              void* d_out, int out_size, void* d_ws, size_t ws_size,
                              hipStream_t stream) {
  (void)in_sizes; (void)n_in; (void)out_size; (void)ws_size;
  const float* x    = (const float*)d_in[0];
  const float* gn_w = (const float*)d_in[1];
  const float* gn_b = (const float*)d_in[2];
  const float* q_w  = (const float*)d_in[3];
  const float* q_b  = (const float*)d_in[4];
  const float* k_w  = (const float*)d_in[5];
  const float* k_b  = (const float*)d_in[6];
  const float* v_w  = (const float*)d_in[7];
  const float* v_b  = (const float*)d_in[8];
  const float* p_w  = (const float*)d_in[9];
  const float* p_b  = (const float*)d_in[10];
  float* out = (float*)d_out;

  char* ws = (char*)d_ws;
  float* stats       = (float*)ws;                    // 512 f32
  unsigned short* qb = (unsigned short*)(ws + 4096);  // [B,N,16] bf16
  unsigned short* kb = qb + (size_t)B_ * N_ * D_;     // [B,N,16] bf16
  unsigned short* vb = kb + (size_t)B_ * N_ * D_;     // [B,C,N]  bf16
  unsigned short* ab = vb + (size_t)B_ * N_ * C_;     // [B,N,C]  bf16 (a^T)

  hipLaunchKernelGGL(gn_stats_kernel, dim3(B_ * 32), dim3(256), 0, stream, x, stats);
  hipLaunchKernelGGL(qkv_kernel, dim3(B_ * 64), dim3(256), 0, stream, x, stats, gn_w,
                     gn_b, q_w, q_b, k_w, k_b, v_w, v_b, qb, kb, vb);
  hipLaunchKernelGGL(attn_kernel, dim3(B_ * 32), dim3(256), 0, stream, qb, kb, vb, ab);
  hipLaunchKernelGGL(proj_kernel, dim3(B_ * 64), dim3(256), 0, stream, x, p_w, p_b, ab,
                     out);
}
